// CorrelationConcat_48292612276566
// MI455X (gfx1250) — matrix operand order repaired and checked by reading, not compiled, not run on hardware
//
#include <hip/hip_runtime.h>

// Problem constants (match reference)
#define BB  32
#define CC  256
#define KK  64
#define HH  64
#define WW  64
#define NN  4096   // H*W
#define INC 320    // C + K

typedef __attribute__((ext_vector_type(16))) __bf16 v16bf;
typedef __attribute__((ext_vector_type(8)))  float  v8f;

struct __attribute__((aligned(16))) U128 { unsigned int x, y, z, w; };

union BF16Frag {
    v16bf          v;
    unsigned short s[16];
    U128           q[2];
};

// round-to-nearest-even f32 -> bf16 (raw u16)
__device__ __forceinline__ unsigned short f2bf(float f) {
    unsigned int u = __float_as_uint(f);
    u += 0x7FFFu + ((u >> 16) & 1u);
    return (unsigned short)(u >> 16);
}

// ---------------------------------------------------------------------------
// WMMA fragment loaders (ISA 7.12.2 layouts, wave32). Both operands are stored
// K-contiguous per lane, so every load is a 16-byte vector load.
//
// A: 16x32 bf16, row-major [M,K], lda = row stride.
//    lane<16: row=lane,    elems 0-7 = K k0+0..7,  elems 8-15 = K k0+16..23
//    lane>=16: row=lane-16, elems 0-7 = K k0+8..15, elems 8-15 = K k0+24..31
__device__ __forceinline__ BF16Frag load_a_frag(const unsigned short* __restrict__ A,
                                                int lda, int row0, int k0, int lane) {
    int row = row0 + (lane & 15);
    int kb  = k0 + ((lane >> 4) << 3);
    const unsigned short* p = A + (long)row * lda + kb;
    BF16Frag f;
    f.q[0] = *(const U128*)(p);        // K kb..kb+7
    f.q[1] = *(const U128*)(p + 16);   // K kb+16..kb+23
    return f;
}

// B: 32x16 bf16. lane&15 = column; same K order as A.
//    elems 0-7 = K kb..kb+7, elems 8-15 = K kb+16..kb+23, kb = k0 + ((lane>>4)<<3)
//    memory is K-contiguous: elem (kk,col) at B + col*ldbn + kk
__device__ __forceinline__ BF16Frag load_b_frag(const unsigned short* __restrict__ Bm,
                                                int ldbn, int k0, int col0, int lane) {
    int col = col0 + (lane & 15);
    int kb  = k0 + ((lane >> 4) << 3);
    const unsigned short* p = Bm + (long)col * ldbn + kb;
    BF16Frag f;
    f.q[0] = *(const U128*)(p);        // K kb..kb+7
    f.q[1] = *(const U128*)(p + 16);   // K kb+16..kb+23
    return f;
}

__device__ __forceinline__ v8f wmma_bf16(const BF16Frag& a, const BF16Frag& b, v8f c) {
    return __builtin_amdgcn_wmma_f32_16x16x32_bf16(false, a.v, false, b.v,
                                                   (short)0, c, false, false);
}

// ---------------------------------------------------------------------------
// Generic batched WMMA GEMM:  O[b] = epilogue( A[b] (MxKin) * B[b] (Kin x Ncols) )
// B[b] is stored K-contiguous: element (kk,col) at Bm + col*ldbn + kk.
// Block: 256 threads = 8 waves arranged 2(M) x 4(N).
// Wave tile: 32(M) x 16*NT(N);  block tile: 64(M) x 64*NT(N).
// All launched shapes divide exactly.
// EPI: 0 = f32 [M,ldo]
//      1 = +bias, BN, ReLU -> f32 [M,ldo]  AND  bf16 transposed [Ncols,ldt]
//      2 = +bias -> bf16 [M,ldo]
//      3 = *scale -> f32 [M,ldo]
//      4 = bf16 [M,ldo]
//      5 = +bias +residual -> f32 [M,ldo]
//      6 = +bias -> bf16 transposed [Ncols,ldt]
template <int EPI, int NT>
__global__ __launch_bounds__(256)
void wmma_gemm(const unsigned short* __restrict__ A, long sAb, int lda,
               const unsigned short* __restrict__ Bm, long sBb, int ldbn,
               float* __restrict__ Of, long sOf,
               unsigned short* __restrict__ Ob, long sOb,
               int M, int Ncols, int Kin, int ldo, int ldt, float scale,
               const float* __restrict__ bias,
               const float* __restrict__ bn_mean, const float* __restrict__ bn_var,
               const float* __restrict__ bn_gamma, const float* __restrict__ bn_beta,
               const float* __restrict__ addF, long sAdd)
{
    const int lane = threadIdx.x & 31;
    const int wave = threadIdx.x >> 5;
    const int b    = blockIdx.z;
    const int m0   = blockIdx.y * 64 + (wave & 1) * 32;
    const int n0   = (blockIdx.x * 4 + (wave >> 1)) * (16 * NT);

    const unsigned short* Ab = A  + (long)b * sAb;
    const unsigned short* Bb = Bm + (long)b * sBb;

    v8f acc[2][NT];
    v8f zer = {0.f, 0.f, 0.f, 0.f, 0.f, 0.f, 0.f, 0.f};
#pragma unroll
    for (int i = 0; i < 2; ++i)
#pragma unroll
    for (int j = 0; j < NT; ++j) acc[i][j] = zer;

    for (int k0 = 0; k0 < Kin; k0 += 32) {
        BF16Frag a0 = load_a_frag(Ab, lda, m0,      k0, lane);
        BF16Frag a1 = load_a_frag(Ab, lda, m0 + 16, k0, lane);
        BF16Frag bfr[NT];
#pragma unroll
        for (int j = 0; j < NT; ++j)
            bfr[j] = load_b_frag(Bb, ldbn, k0, n0 + 16 * j, lane);
        if (k0 + 32 < Kin) {   // gfx1250 global_prefetch_b8, WGP scope (locality 3)
            __builtin_prefetch(Ab + (long)(m0 + (lane & 15)) * lda + k0 + 32, 0, 3);
            __builtin_prefetch(Bb + (long)(n0 + (lane & 15)) * ldbn + k0 + 32, 0, 3);
        }
#pragma unroll
        for (int j = 0; j < NT; ++j) {
            acc[0][j] = wmma_bf16(a0, bfr[j], acc[0][j]);
            acc[1][j] = wmma_bf16(a1, bfr[j], acc[1][j]);
        }
    }

#pragma unroll
    for (int mt = 0; mt < 2; ++mt)
#pragma unroll
    for (int nt = 0; nt < NT; ++nt)
#pragma unroll
    for (int r = 0; r < 8; ++r) {
        int row = m0 + mt * 16 + r + ((lane >> 4) << 3);
        int col = n0 + nt * 16 + (lane & 15);
        float v = acc[mt][nt][r];
        long  o = (long)row * ldo + col;
        if constexpr (EPI == 0) {
            Of[(long)b * sOf + o] = v;
        } else if constexpr (EPI == 1) {
            v += bias[row];
            v  = (v - bn_mean[row]) * rsqrtf(bn_var[row] + 1e-5f);
            v  = fmaxf(v * bn_gamma[row] + bn_beta[row], 0.f);
            Of[(long)b * sOf + o] = v;
            Ob[(long)b * sOb + (long)col * ldt + row] = f2bf(v);   // [N, C]
        } else if constexpr (EPI == 2) {
            Ob[(long)b * sOb + o] = f2bf(v + bias[row]);
        } else if constexpr (EPI == 3) {
            Of[(long)b * sOf + o] = v * scale;
        } else if constexpr (EPI == 4) {
            Ob[(long)b * sOb + o] = f2bf(v);
        } else if constexpr (EPI == 5) {
            Of[(long)b * sOf + o] = v + bias[row] + addF[(long)b * sAdd + o];
        } else {  // EPI == 6
            Ob[(long)b * sOb + (long)col * ldt + row] = f2bf(v + bias[row]);
        }
    }
}

// ---------------------------------------------------------------------------
// Elementwise / helper kernels
__global__ void cast_kernel(const float* __restrict__ src, unsigned short* __restrict__ dst,
                            long count) {
    long i = (long)blockIdx.x * blockDim.x + threadIdx.x;
    if (i < count) dst[i] = f2bf(src[i]);
}

// x[b,c,n] -> xT_bf16[b,n,c]  (correlation B operand: K=c contiguous)
__global__ void xT_kernel(const float* __restrict__ x, unsigned short* __restrict__ xT) {
    long i = (long)blockIdx.x * blockDim.x + threadIdx.x;  // B*C*N, n fastest
    int  n = (int)(i & (NN - 1));
    long t = i >> 12;
    int  c = (int)(t % CC);
    int  b = (int)(t / CC);
    xT[((long)b * NN + n) * CC + c] = f2bf(x[i]);
}

// z[b,c,k] -> zT_bf16[b,k,c]  (correlation A operand row-major)
__global__ void zT_kernel(const float* __restrict__ z, unsigned short* __restrict__ zT) {
    long i = (long)blockIdx.x * blockDim.x + threadIdx.x;
    if (i >= (long)BB * CC * KK) return;
    int k = (int)(i % KK);
    long t = i / KK;
    int c = (int)(t % CC);
    int b = (int)(t / CC);
    zT[((long)b * KK + k) * CC + c] = f2bf(z[i]);
}

// W = wo (CxC) @ wi (CxC), bf16 out (folds 1x1-proj into q/k/v weights)
__global__ void combine_w_kernel(const float* __restrict__ wo, const float* __restrict__ wi,
                                 unsigned short* __restrict__ out) {
    int idx = blockIdx.x * 256 + threadIdx.x;  // C*C threads
    int o = idx >> 8, i = idx & 255;
    float acc = 0.f;
    for (int c = 0; c < CC; ++c) acc += wo[o * CC + c] * wi[c * CC + i];
    out[idx] = f2bf(acc);
}

// bias = wo @ inp_b + ob
__global__ void combine_b_kernel(const float* __restrict__ wo, const float* __restrict__ inb,
                                 const float* __restrict__ ob, float* __restrict__ out) {
    int o = threadIdx.x;  // 256
    float acc = ob[o];
    for (int c = 0; c < CC; ++c) acc += wo[o * CC + c] * inb[c];
    out[o] = acc;
}

// depthwise 3x3 (pad 1) over concat([corr(K ch), d(C ch)]) + bias
// -> bf16, PIXEL-MAJOR [b][n][ch] so the pointwise GEMM B operand is contiguous
__global__ void depthwise_kernel(const float* __restrict__ corr, const float* __restrict__ dd,
                                 const float* __restrict__ dw_w, const float* __restrict__ dw_b,
                                 unsigned short* __restrict__ out) {
    long idx = (long)blockIdx.x * blockDim.x + threadIdx.x;  // B*INC*N, n fastest
    int  n   = (int)(idx & (NN - 1));
    long t   = idx >> 12;
    int  ch  = (int)(t % INC);
    int  b   = (int)(t / INC);
    int  xx  = n & (WW - 1), yy = n >> 6;
    const float* src = (ch < KK) ? corr + ((long)b * KK + ch) * NN
                                 : dd   + ((long)b * CC + (ch - KK)) * NN;
    const float* wp = dw_w + ch * 9;
    float acc = dw_b[ch];
#pragma unroll
    for (int dy = -1; dy <= 1; ++dy) {
        int y2 = yy + dy;
        if ((unsigned)y2 >= HH) continue;
#pragma unroll
        for (int dx = -1; dx <= 1; ++dx) {
            int x2 = xx + dx;
            if ((unsigned)x2 >= WW) continue;
            acc += src[y2 * WW + x2] * wp[(dy + 1) * 3 + (dx + 1)];
        }
    }
    out[((long)b * NN + n) * INC + ch] = f2bf(acc);
}

// softmax along last dim of score[b,c, :CC]; wave32 per row; f32 + bf16 out
__global__ __launch_bounds__(256)
void softmax_kernel(float* __restrict__ score, unsigned short* __restrict__ attb) {
    int row  = blockIdx.x * 8 + (threadIdx.x >> 5);   // B*C rows
    int lane = threadIdx.x & 31;
    float* p = score + (long)row * CC;
    float vals[8];
    float mx = -3.4e38f;
#pragma unroll
    for (int i = 0; i < 8; ++i) { vals[i] = p[lane + 32 * i]; mx = fmaxf(mx, vals[i]); }
#pragma unroll
    for (int off = 16; off; off >>= 1) mx = fmaxf(mx, __shfl_xor(mx, off, 32));
    float sum = 0.f;
#pragma unroll
    for (int i = 0; i < 8; ++i) { vals[i] = __expf(vals[i] - mx); sum += vals[i]; }
#pragma unroll
    for (int off = 16; off; off >>= 1) sum += __shfl_xor(sum, off, 32);
    float inv = 1.f / sum;
#pragma unroll
    for (int i = 0; i < 8; ++i) {
        float v = vals[i] * inv;
        p[lane + 32 * i] = v;
        attb[(long)row * CC + lane + 32 * i] = f2bf(v);
    }
}

// ---------------------------------------------------------------------------
extern "C" void kernel_launch(void* const* d_in, const int* in_sizes, int n_in,
                              void* d_out, int out_size, void* d_ws, size_t ws_size,
                              hipStream_t stream) {
    const float* z    = (const float*)d_in[0];
    const float* x    = (const float*)d_in[1];
    const float* d    = (const float*)d_in[2];
    const float* dw_w = (const float*)d_in[3];
    const float* dw_b = (const float*)d_in[4];
    const float* pw_w = (const float*)d_in[5];
    const float* pw_b = (const float*)d_in[6];
    const float* bn_g = (const float*)d_in[7];
    const float* bn_b = (const float*)d_in[8];
    const float* bn_m = (const float*)d_in[9];
    const float* bn_v = (const float*)d_in[10];
    const float* inp_w = (const float*)d_in[11];
    const float* inp_b = (const float*)d_in[12];
    const float* q_w = (const float*)d_in[13];
    const float* q_b = (const float*)d_in[14];
    const float* k_w = (const float*)d_in[15];
    const float* k_b = (const float*)d_in[16];
    const float* v_w = (const float*)d_in[17];
    const float* v_b = (const float*)d_in[18];
    const float* op_w = (const float*)d_in[19];
    const float* op_b = (const float*)d_in[20];
    float* out = (float*)d_out;

    char*  ws  = (char*)d_ws;
    size_t off = 0;
    auto alloc = [&](size_t bytes) -> void* {
        void* p = ws + off;
        off = (off + bytes + 255) & ~(size_t)255;
        return p;
    };

    const long BCN = (long)BB * CC * NN;
    unsigned short* xT_bf  = (unsigned short*)alloc(BCN * 2);                 // [B][N][C]
    unsigned short* zT_bf  = (unsigned short*)alloc((long)BB * KK * CC * 2);  // [B][K][C]
    unsigned short* pww_bf = (unsigned short*)alloc((long)CC * INC * 2);
    unsigned short* opw_bf = (unsigned short*)alloc((long)CC * CC * 2);
    unsigned short* Wq_bf  = (unsigned short*)alloc((long)CC * CC * 2);
    unsigned short* Wk_bf  = (unsigned short*)alloc((long)CC * CC * 2);
    unsigned short* Wv_bf  = (unsigned short*)alloc((long)CC * CC * 2);
    float* bq = (float*)alloc(CC * 4);
    float* bk = (float*)alloc(CC * 4);
    float* bv = (float*)alloc(CC * 4);
    float*          corr  = (float*)alloc((long)BB * KK * NN * 4);            // [B][K][N]
    unsigned short* dwT   = (unsigned short*)alloc((long)BB * NN * INC * 2);  // [B][N][IN]
    float*          s_f   = (float*)alloc(BCN * 4);                           // [B][C][N]
    unsigned short* sT_bf = (unsigned short*)alloc(BCN * 2);                  // [B][N][C]
    unsigned short* qbf   = (unsigned short*)alloc(BCN * 2);                  // [B][C][N]
    unsigned short* kbf   = (unsigned short*)alloc(BCN * 2);                  // [B][C][N]
    unsigned short* vT_bf = (unsigned short*)alloc(BCN * 2);                  // [B][N][C]
    float*          score = (float*)alloc((long)BB * CC * CC * 4);
    unsigned short* attbf = (unsigned short*)alloc((long)BB * CC * CC * 2);
    unsigned short* M2bf  = (unsigned short*)alloc((long)BB * CC * CC * 2);

    // ---- stage 0: casts + weight folding ----
    xT_kernel<<<(unsigned)(BCN / 256), 256, 0, stream>>>(x, xT_bf);
    zT_kernel<<<(unsigned)(((long)BB * CC * KK + 255) / 256), 256, 0, stream>>>(z, zT_bf);
    cast_kernel<<<(CC * INC + 255) / 256, 256, 0, stream>>>(pw_w, pww_bf, (long)CC * INC);
    cast_kernel<<<(CC * CC + 255) / 256, 256, 0, stream>>>(op_w, opw_bf, (long)CC * CC);
    combine_w_kernel<<<CC * CC / 256, 256, 0, stream>>>(q_w, inp_w, Wq_bf);
    combine_w_kernel<<<CC * CC / 256, 256, 0, stream>>>(k_w, inp_w, Wk_bf);
    combine_w_kernel<<<CC * CC / 256, 256, 0, stream>>>(v_w, inp_w, Wv_bf);
    combine_b_kernel<<<1, CC, 0, stream>>>(q_w, inp_b, q_b, bq);
    combine_b_kernel<<<1, CC, 0, stream>>>(k_w, inp_b, k_b, bk);
    combine_b_kernel<<<1, CC, 0, stream>>>(v_w, inp_b, v_b, bv);

    // ---- stage 1: correlation  corr[b] = zT[b] (64x256) * x[b] (256x4096) ----
    wmma_gemm<0, 4><<<dim3(NN / 256, KK / 64, BB), 256, 0, stream>>>(
        zT_bf, (long)KK * CC, CC, xT_bf, BCN / BB, CC,
        corr, (long)KK * NN, nullptr, 0, KK, NN, CC, NN, 0, 1.f,
        nullptr, nullptr, nullptr, nullptr, nullptr, nullptr, 0);

    // ---- stage 2: depthwise 3x3 on [corr ; d] -> dwT (bf16, pixel-major) ----
    depthwise_kernel<<<(unsigned)(((long)BB * INC * NN) / 256), 256, 0, stream>>>(
        corr, d, dw_w, dw_b, dwT);

    // ---- stage 3: pointwise + BN + ReLU: s = relu(bn(pw_w*dw + pw_b)) ----
    wmma_gemm<1, 4><<<dim3(NN / 256, CC / 64, BB), 256, 0, stream>>>(
        pww_bf, 0, INC, dwT, (long)NN * INC, INC,
        s_f, BCN / BB, sT_bf, BCN / BB, CC, NN, INC, NN, CC, 1.f,
        pw_b, bn_m, bn_v, bn_g, bn_b, nullptr, 0);

    // ---- stage 4: q/k/v = Wx*s + bx  (inp 1x1 folded in) ----
    wmma_gemm<2, 4><<<dim3(NN / 256, CC / 64, BB), 256, 0, stream>>>(
        Wq_bf, 0, CC, sT_bf, BCN / BB, CC,
        nullptr, 0, qbf, BCN / BB, CC, NN, CC, NN, 0, 1.f,
        bq, nullptr, nullptr, nullptr, nullptr, nullptr, 0);
    wmma_gemm<2, 4><<<dim3(NN / 256, CC / 64, BB), 256, 0, stream>>>(
        Wk_bf, 0, CC, sT_bf, BCN / BB, CC,
        nullptr, 0, kbf, BCN / BB, CC, NN, CC, NN, 0, 1.f,
        bk, nullptr, nullptr, nullptr, nullptr, nullptr, 0);
    wmma_gemm<6, 4><<<dim3(NN / 256, CC / 64, BB), 256, 0, stream>>>(
        Wv_bf, 0, CC, sT_bf, BCN / BB, CC,
        nullptr, 0, vT_bf, BCN / BB, CC, NN, CC, NN, CC, 1.f,
        bv, nullptr, nullptr, nullptr, nullptr, nullptr, 0);

    // ---- stage 5: score[b] = q[b] * k[b]^T / sqrt(C)  (K = 4096) ----
    wmma_gemm<3, 2><<<dim3(CC / 128, CC / 64, BB), 256, 0, stream>>>(
        qbf, BCN / BB, NN, kbf, BCN / BB, NN,
        score, (long)CC * CC, nullptr, 0, CC, CC, NN, CC, 0, 0.0625f,
        nullptr, nullptr, nullptr, nullptr, nullptr, nullptr, 0);

    // ---- stage 6: softmax rows -> att (f32 in place) + att_bf16 ----
    softmax_kernel<<<BB * CC / 8, 256, 0, stream>>>(score, attbf);

    // ---- stage 7: M2[b] = op_w * att[b]^T  (fold output proj into att) ----
    wmma_gemm<4, 2><<<dim3(CC / 128, CC / 64, BB), 256, 0, stream>>>(
        opw_bf, 0, CC, attbf, (long)CC * CC, CC,
        nullptr, 0, M2bf, (long)CC * CC, CC, CC, CC, CC, 0, 1.f,
        nullptr, nullptr, nullptr, nullptr, nullptr, nullptr, 0);

    // ---- stage 8: out = M2[b] * v[b] + op_b + s ----
    wmma_gemm<5, 4><<<dim3(NN / 256, CC / 64, BB), 256, 0, stream>>>(
        M2bf, (long)CC * CC, CC, vT_bf, BCN / BB, CC,
        out, BCN / BB, nullptr, 0, CC, NN, CC, NN, 0, 1.f,
        op_b, nullptr, nullptr, nullptr, nullptr, s_f, BCN / BB);

    (void)in_sizes; (void)n_in; (void)out_size; (void)ws_size;
}
